// Attention_48842368090493
// MI455X (gfx1250) — matrix operand order repaired and checked by reading, not compiled, not run on hardware
//
#include <hip/hip_runtime.h>

typedef __bf16 bf16;
typedef __attribute__((ext_vector_type(16))) __bf16 v16bf;
typedef __attribute__((ext_vector_type(8)))  __bf16 v8bf;
typedef __attribute__((ext_vector_type(8)))  float  v8f;

#define DIM    768
#define NSEQ   1024
#define BATCH  16
#define NHEADS 12
#define HD     64
#define MROWS  (BATCH * NSEQ)          // 16384
#define ATT_SCALE 0.03608439182435161f // 1/sqrt(768) (module scales by sqrt(dim))

// ---------------------------------------------------------------------------
// WMMA helpers (CDNA5 wave32, 16x16x32 bf16 -> f32)
// ---------------------------------------------------------------------------
__device__ __forceinline__ v8f wmma_bf16(v16bf a, v16bf b, v8f c) {
  return __builtin_amdgcn_wmma_f32_16x16x32_bf16(false, a, false, b,
                                                 (short)0, c, false, false);
}

// A-matrix 16x32 bf16, row-major source, ld in elements.
// ISA layout: lane<16 holds row M=lane, k {0..7} in v0..3 and {16..23} in v4..7;
// lane>=16 holds row M=lane-16, k {8..15} and {24..31}.
__device__ __forceinline__ v16bf load_a16x32(const bf16* base, int ld) {
  int lane = threadIdx.x & 31;
  const bf16* p = base + (size_t)(lane & 15) * ld + ((lane & 16) ? 8 : 0);
  v8bf lo = *(const v8bf*)p;          // k = koff .. koff+7
  v8bf hi = *(const v8bf*)(p + 16);   // k = koff+16 .. koff+23
  return __builtin_shufflevector(lo, hi, 0, 1, 2, 3, 4, 5, 6, 7,
                                 8, 9, 10, 11, 12, 13, 14, 15);
}

// B-matrix 32x16 bf16 from a k-major source: element (k,n) at base[n*ld + k].
// ISA layout: lanes 0-15 hold K {0..7,16..23} of column n=lane; lanes 16-31 hold K {8..15,24..31}.
__device__ __forceinline__ v16bf load_b32x16(const bf16* base, int ld) {
  int lane = threadIdx.x & 31;
  const bf16* p = base + (size_t)(lane & 15) * ld + ((lane & 16) ? 8 : 0);
  v8bf lo = *(const v8bf*)p;
  v8bf hi = *(const v8bf*)(p + 16);
  return __builtin_shufflevector(lo, hi, 0, 1, 2, 3, 4, 5, 6, 7,
                                 8, 9, 10, 11, 12, 13, 14, 15);
}

// ---------------------------------------------------------------------------
// Conversion kernels
// ---------------------------------------------------------------------------
__global__ void k_cvt(const float* __restrict__ x, bf16* __restrict__ y, int n) {
  int i = blockIdx.x * blockDim.x + threadIdx.x;
  if (i < n) y[i] = (bf16)x[i];
}

// Transpose 768x768 f32 -> bf16 (Wt[n*DIM + k] = W[k*DIM + n])
__global__ void k_cvt_t(const float* __restrict__ w, bf16* __restrict__ wt) {
  int i = blockIdx.x * blockDim.x + threadIdx.x;
  if (i < DIM * DIM) {
    int n = i / DIM, k = i % DIM;
    wt[i] = (bf16)w[k * DIM + n];
  }
}

// ---------------------------------------------------------------------------
// GEMM: Y[m,n] = sum_k A[m,k]*Bt[n,k] + bias[n]
// A: bf16 [MROWS x DIM] row-major. Bt: bf16 [DIM x DIM] (transposed weight).
// mode 0: bf16 row-major out (ld = DIM)
// mode 1: bf16 transposed-per-batch out: out[(batch*DIM + n)*NSEQ + seq]
// mode 2: f32 row-major out (final projection -> d_out)
// One wave computes a 32x64 tile: 8 accumulators, 192 WMMAs total.
// ---------------------------------------------------------------------------
__global__ void k_gemm(const bf16* __restrict__ A, const bf16* __restrict__ Bt,
                       const float* __restrict__ bias, void* __restrict__ outp,
                       int mode) {
  int wave = blockIdx.x * (blockDim.x >> 5) + (threadIdx.x >> 5);
  int lane = threadIdx.x & 31;
  int m0 = (wave / (DIM / 64)) * 32;
  int n0 = (wave % (DIM / 64)) * 64;

  v8f acc[2][4];
#pragma unroll
  for (int i = 0; i < 2; ++i)
#pragma unroll
    for (int j = 0; j < 4; ++j) acc[i][j] = (v8f)(0.0f);

  for (int kb = 0; kb < DIM; kb += 32) {
    v16bf a0 = load_a16x32(A + (size_t)m0 * DIM + kb, DIM);
    v16bf a1 = load_a16x32(A + (size_t)(m0 + 16) * DIM + kb, DIM);
#pragma unroll
    for (int j = 0; j < 4; ++j) {
      v16bf b = load_b32x16(Bt + (size_t)(n0 + j * 16) * DIM + kb, DIM);
      acc[0][j] = wmma_bf16(a0, b, acc[0][j]);
      acc[1][j] = wmma_bf16(a1, b, acc[1][j]);
    }
  }

  int col = lane & 15;
  int roff = (lane & 16) ? 8 : 0;
#pragma unroll
  for (int j = 0; j < 4; ++j) {
    int n = n0 + j * 16 + col;
    float bv = bias[n];
#pragma unroll
    for (int i = 0; i < 2; ++i) {
#pragma unroll
      for (int r = 0; r < 8; ++r) {
        int m = m0 + i * 16 + roff + r;
        float val = acc[i][j][r] + bv;
        if (mode == 0) {
          ((bf16*)outp)[(size_t)m * DIM + n] = (bf16)val;
        } else if (mode == 1) {
          int bb = m >> 10, s = m & 1023;
          ((bf16*)outp)[((size_t)bb * DIM + n) * NSEQ + s] = (bf16)val;
        } else {
          ((float*)outp)[(size_t)m * DIM + n] = val;
        }
      }
    }
  }
}

// ---------------------------------------------------------------------------
// Flash attention: one wave = one (batch, head, 16-query tile).
// Q held in registers (two 16x32 A tiles); iterate keys in 32-wide blocks:
//   S = Q K^T (4 WMMAs) -> f32 online softmax (half-wave shuffle reductions)
//   P bf16 bounced through LDS (C-layout -> A-layout) -> O += P V (4 WMMAs)
// ---------------------------------------------------------------------------
__global__ void k_attn(const bf16* __restrict__ Q, const bf16* __restrict__ K,
                       const bf16* __restrict__ Vt, bf16* __restrict__ Ctx) {
  __shared__ __align__(16) bf16 plds[4][16][32];
  int wv = threadIdx.x >> 5;
  int lane = threadIdx.x & 31;
  int wave = blockIdx.x * 4 + wv;
  int qt = wave & 63;                 // 64 query tiles of 16
  int h = (wave >> 6) % NHEADS;
  int b = wave / (64 * NHEADS);

  const bf16* Qb = Q + ((size_t)(b * NSEQ + qt * 16)) * DIM + h * HD;
  const bf16* Kb = K + (size_t)b * NSEQ * DIM + h * HD;
  const bf16* Vb = Vt + ((size_t)b * DIM + h * HD) * NSEQ;

  v16bf qa0 = load_a16x32(Qb, DIM);       // d = 0..31
  v16bf qa1 = load_a16x32(Qb + 32, DIM);  // d = 32..63

  float mi[8], li[8];
  v8f o[4];
#pragma unroll
  for (int r = 0; r < 8; ++r) { mi[r] = -__builtin_inff(); li[r] = 0.0f; }
#pragma unroll
  for (int c = 0; c < 4; ++c) o[c] = (v8f)(0.0f);

  int col = lane & 15;
  int roff = (lane & 16) ? 8 : 0;

  for (int kb = 0; kb < NSEQ; kb += 32) {
    // scores for 32 keys (two 16x16 C tiles)
    v8f s0 = (v8f)(0.0f), s1 = (v8f)(0.0f);
    {
      v16bf b0 = load_b32x16(Kb + (size_t)kb * DIM, DIM);
      v16bf b1 = load_b32x16(Kb + (size_t)kb * DIM + 32, DIM);
      s0 = wmma_bf16(qa0, b0, s0);
      s0 = wmma_bf16(qa1, b1, s0);
    }
    {
      v16bf b0 = load_b32x16(Kb + (size_t)(kb + 16) * DIM, DIM);
      v16bf b1 = load_b32x16(Kb + (size_t)(kb + 16) * DIM + 32, DIM);
      s1 = wmma_bf16(qa0, b0, s1);
      s1 = wmma_bf16(qa1, b1, s1);
    }

    // online softmax (rows live across half-wave lanes; reduce over cols)
    float p0[8], p1[8], corr[8];
#pragma unroll
    for (int r = 0; r < 8; ++r) {
      float x0 = s0[r] * ATT_SCALE;
      float x1 = s1[r] * ATT_SCALE;
      float rm = fmaxf(x0, x1);
      rm = fmaxf(rm, __shfl_xor(rm, 1, 32));
      rm = fmaxf(rm, __shfl_xor(rm, 2, 32));
      rm = fmaxf(rm, __shfl_xor(rm, 4, 32));
      rm = fmaxf(rm, __shfl_xor(rm, 8, 32));
      float mn = fmaxf(mi[r], rm);
      float c = __expf(mi[r] - mn);
      float e0 = __expf(x0 - mn);
      float e1 = __expf(x1 - mn);
      float rs = e0 + e1;
      rs += __shfl_xor(rs, 1, 32);
      rs += __shfl_xor(rs, 2, 32);
      rs += __shfl_xor(rs, 4, 32);
      rs += __shfl_xor(rs, 8, 32);
      li[r] = li[r] * c + rs;
      mi[r] = mn;
      corr[r] = c;
      p0[r] = e0;
      p1[r] = e1;
    }

    // rescale running context
#pragma unroll
    for (int c = 0; c < 4; ++c)
#pragma unroll
      for (int r = 0; r < 8; ++r) o[c][r] *= corr[r];

    // C-layout -> A-layout via per-wave LDS tile (same-wave DS ops are in-order)
#pragma unroll
    for (int r = 0; r < 8; ++r) {
      plds[wv][roff + r][col] = (bf16)p0[r];
      plds[wv][roff + r][col + 16] = (bf16)p1[r];
    }
    v16bf pa = load_a16x32(&plds[wv][0][0], 32);

    // O += P * V  (V stored transposed: element (k,n) at Vb[n*NSEQ + k])
#pragma unroll
    for (int c = 0; c < 4; ++c) {
      v16bf vb = load_b32x16(Vb + (size_t)(c * 16) * NSEQ + kb, NSEQ);
      o[c] = wmma_bf16(pa, vb, o[c]);
    }
  }

  // normalize + store context (bf16, row-major, feeds final projection as A)
#pragma unroll
  for (int r = 0; r < 8; ++r) {
    float inv = 1.0f / li[r];
    size_t row = (size_t)b * NSEQ + qt * 16 + roff + r;
#pragma unroll
    for (int c = 0; c < 4; ++c)
      Ctx[row * DIM + h * HD + c * 16 + col] = (bf16)(o[c][r] * inv);
  }
}

// ---------------------------------------------------------------------------
extern "C" void kernel_launch(void* const* d_in, const int* in_sizes, int n_in,
                              void* d_out, int out_size, void* d_ws, size_t ws_size,
                              hipStream_t stream) {
  const float* X  = (const float*)d_in[0];
  const float* Wq = (const float*)d_in[1];
  const float* bq = (const float*)d_in[2];
  const float* Wk = (const float*)d_in[3];
  const float* bk = (const float*)d_in[4];
  const float* Wv = (const float*)d_in[5];
  const float* bv = (const float*)d_in[6];
  const float* Wo = (const float*)d_in[7];
  const float* bo = (const float*)d_in[8];

  bf16* Xb  = (bf16*)d_ws;
  bf16* WqT = Xb + (size_t)MROWS * DIM;
  bf16* WkT = WqT + (size_t)DIM * DIM;
  bf16* WvT = WkT + (size_t)DIM * DIM;
  bf16* WoT = WvT + (size_t)DIM * DIM;
  bf16* Qb  = WoT + (size_t)DIM * DIM;
  bf16* Kb  = Qb + (size_t)MROWS * DIM;
  bf16* Vt  = Kb + (size_t)MROWS * DIM;
  bf16* Cx  = Vt + (size_t)MROWS * DIM;

  const int NX = MROWS * DIM;   // 12,582,912
  const int NW = DIM * DIM;     // 589,824

  k_cvt<<<(NX + 255) / 256, 256, 0, stream>>>(X, Xb, NX);
  k_cvt_t<<<(NW + 255) / 256, 256, 0, stream>>>(Wq, WqT);
  k_cvt_t<<<(NW + 255) / 256, 256, 0, stream>>>(Wk, WkT);
  k_cvt_t<<<(NW + 255) / 256, 256, 0, stream>>>(Wv, WvT);
  k_cvt_t<<<(NW + 255) / 256, 256, 0, stream>>>(Wo, WoT);

  // 16384/32 * 768/64 = 6144 waves; 4 waves per 128-thread block
  dim3 gB(6144 / 4), tB(128);
  k_gemm<<<gB, tB, 0, stream>>>(Xb, WqT, bq, (void*)Qb, 0);
  k_gemm<<<gB, tB, 0, stream>>>(Xb, WkT, bk, (void*)Kb, 0);
  k_gemm<<<gB, tB, 0, stream>>>(Xb, WvT, bv, (void*)Vt, 1);  // V transposed

  // 16 * 12 * 64 = 12288 waves
  k_attn<<<12288 / 4, 128, 0, stream>>>(Qb, Kb, Vt, Cx);

  k_gemm<<<gB, tB, 0, stream>>>(Cx, WoT, bo, d_out, 2);      // f32 output
}
